// Decoder_10393820856362
// MI455X (gfx1250) — matrix operand order repaired and checked by reading, not compiled, not run on hardware
//
#include <hip/hip_runtime.h>
#include <hip/hip_bf16.h>
#include <math.h>

// ---------------------------------------------------------------------------
// Transformer decoder forward (S=50, B=1024, D=256, H=8, HD=32, FF=512, L=4)
// for MI455X / gfx1250.
//   * All big matmuls: v_wmma_f32_16x16x32_f16, 128x64 block tile, 32x32 wave
//     tile (4 WMMA/wave/K-step), double-buffered LDS, 1 barrier per K-step.
//   * GEMM operands live in memory as f16 (produced by epilogue kernels and a
//     one-time weight transpose-convert), so LDS staging is a pure byte copy:
//     GLOBAL_LOAD_ASYNC_TO_LDS_B128 (builtin confirmed present; takes
//     (i32x4 AS1*, i32x4 AS3*, imm, imm)), retired with s_wait_asynccnt.
//   * fp32 kept for the residual stream, softmax and LN statistics.
// ---------------------------------------------------------------------------

constexpr int S_  = 50;
constexpr int B_  = 1024;
constexpr int D_  = 256;
constexpr int H_  = 8;
constexpr int HD_ = 32;
constexpr int FF_ = 512;
constexpr int E_  = 64;
constexpr int L_  = 4;
constexpr int T_  = S_ * B_;     // 51200 tokens, token t = s*B + b  ([S,B,D])

typedef _Float16 v16h __attribute__((ext_vector_type(16)));
typedef _Float16 h8v  __attribute__((ext_vector_type(8)));
typedef _Float16 h4v  __attribute__((ext_vector_type(4)));
typedef float    v8f  __attribute__((ext_vector_type(8)));
typedef int      i32x4 __attribute__((vector_size(16)));   // matches builtin

#if defined(__has_builtin)
#if __has_builtin(__builtin_amdgcn_global_load_async_to_lds_b128) && \
    __has_builtin(__builtin_amdgcn_s_wait_asynccnt)
#define USE_ASYNC_LDS 1
#endif
#endif
#ifndef USE_ASYNC_LDS
#define USE_ASYNC_LDS 0
#endif

// ---------------------------------------------------------------------------
// GEMM: out[M,N] = epilogue(A[M,K] @ Bt[N,K]^T + bias[N])
//   A  : f16 row-major [M,K]   (activations, K in {64,256,512})
//   Bt : f16 row-major [N,K]   (pre-transposed weights)
// OM=0: C (f32) = result + bias.   OM=1: Ch (f16) = relu(result + bias).
// Block 256 threads = 8 waves in 4(M) x 2(N) grid of 32x32 wave tiles.
// Requires M%128==0, N%64==0, K%32==0 (true for all calls).
// LDS tiles (double buffered) keep 40-half row stride so all fragment
// ds_load_b128 stay 16B-aligned; staging chunks are 16B-aligned too.
// ---------------------------------------------------------------------------
constexpr int BM = 128, BN = 64, BK = 32;
constexpr int LDA = 40, LDB = 40;
constexpr int ASZ = BM * LDA;   // 5120 halfs
constexpr int BSZ = BN * LDB;   // 2560 halfs

__device__ __forceinline__ void chunk16(const _Float16* g, _Float16* l) {
#if USE_ASYNC_LDS
  __builtin_amdgcn_global_load_async_to_lds_b128(
      (__attribute__((address_space(1))) i32x4*)g,
      (__attribute__((address_space(3))) i32x4*)l, 0, 0);
#else
  *(h8v*)l = *(const h8v*)g;
#endif
}
__device__ __forceinline__ void stage_wait() {
#if USE_ASYNC_LDS
  __builtin_amdgcn_s_wait_asynccnt(0);
#endif
}

template <int OM>
__global__ __launch_bounds__(256) void gemm_wmma_k(
    const _Float16* __restrict__ A, const _Float16* __restrict__ Bt,
    const float* __restrict__ bias, float* __restrict__ C,
    _Float16* __restrict__ Ch, int M, int N, int K) {
  (void)M;
  __shared__ _Float16 As[2 * ASZ];
  __shared__ _Float16 Bs[2 * BSZ];
  const int tid = threadIdx.x;
  const int m0 = blockIdx.y * BM;
  const int n0 = blockIdx.x * BN;
  const int w  = tid >> 5, l = tid & 31;
  const int wm = (w >> 1) * 32;      // wave tile origin in M (0,32,64,96)
  const int wn = (w & 1) * 32;       // wave tile origin in N (0,32)
  const int g  = l >> 4;             // lane group (0: lanes 0-15, 1: 16-31)
  const int ln = l & 15;

  // Chunk (16B) mapping: A panel 128x32 halfs = 512 chunks (2/thread),
  // B panel 64x32 halfs = 256 chunks (1/thread).
  const int ar0 = tid >> 2, ac = (tid & 3) * 8;      // A rows 0..63 / 64..127
  const int bn  = tid >> 2;                          // B row (n) 0..63

  v8f acc[2][2];
#pragma unroll
  for (int mi = 0; mi < 2; ++mi)
#pragma unroll
    for (int ni = 0; ni < 2; ++ni)
#pragma unroll
      for (int r = 0; r < 8; ++r) acc[mi][ni][r] = 0.f;

  auto stage = [&](int k0, int buf) {
    _Float16* Ad = As + buf * ASZ;
    _Float16* Bd = Bs + buf * BSZ;
    const _Float16* g0 = A + (size_t)(m0 + ar0) * K + k0 + ac;
    const _Float16* g1 = A + (size_t)(m0 + ar0 + 64) * K + k0 + ac;
    const _Float16* g2 = Bt + (size_t)(n0 + bn) * K + k0 + ac;
    _Float16* l0 = Ad + ar0 * LDA + ac;
    _Float16* l1 = Ad + (ar0 + 64) * LDA + ac;
    _Float16* l2 = Bd + bn * LDB + ac;
#if USE_ASYNC_LDS
    chunk16(g0, l0); chunk16(g1, l1); chunk16(g2, l2);
#else
    h8v t0 = *(const h8v*)g0;      // split-phase: 3 clause-able loads,
    h8v t1 = *(const h8v*)g1;      // then 3 ds_store_b128
    h8v t2 = *(const h8v*)g2;
    *(h8v*)l0 = t0; *(h8v*)l1 = t1; *(h8v*)l2 = t2;
#endif
  };

  auto compute = [&](int buf) {
    const _Float16* Ab = As + buf * ASZ;
    const _Float16* Bb = Bs + buf * BSZ;
    v16h af[2], bf[2];
#pragma unroll
    for (int mi = 0; mi < 2; ++mi) {
      const _Float16* ap = Ab + (wm + 16 * mi + ln) * LDA;
      h8v alo = *(const h8v*)(ap + 8 * g);        // K = 8g .. 8g+7
      h8v ahi = *(const h8v*)(ap + 16 + 8 * g);   // K = 16+8g .. 16+8g+7
#pragma unroll
      for (int i = 0; i < 8; ++i) { af[mi][i] = alo[i]; af[mi][i + 8] = ahi[i]; }
    }
#pragma unroll
    for (int ni = 0; ni < 2; ++ni) {
      const _Float16* bp = Bb + (wn + 16 * ni + ln) * LDB;
      h8v blo = *(const h8v*)(bp + 8 * g);        // K = 8g .. 8g+7
      h8v bhi = *(const h8v*)(bp + 16 + 8 * g);   // K = 16+8g .. 16+8g+7
#pragma unroll
      for (int i = 0; i < 8; ++i) { bf[ni][i] = blo[i]; bf[ni][i + 8] = bhi[i]; }
    }
#pragma unroll
    for (int mi = 0; mi < 2; ++mi)
#pragma unroll
      for (int ni = 0; ni < 2; ++ni)
        acc[mi][ni] = __builtin_amdgcn_wmma_f32_16x16x32_f16(
            false, af[mi], false, bf[ni], (short)0, acc[mi][ni], false, false);
  };

  stage(0, 0);
  stage_wait();
  __syncthreads();
  int buf = 0;
  for (int k0 = 0; k0 < K; k0 += BK) {
    if (k0 + BK < K) stage(k0 + BK, buf ^ 1);        // overlaps with WMMA
    if (k0 + 2 * BK < K)                             // global_prefetch_b8
      __builtin_prefetch(A + (size_t)(m0 + (tid & 127)) * K + k0 + 2 * BK, 0, 3);
    compute(buf);
    stage_wait();
    __syncthreads();
    buf ^= 1;
  }

  // Epilogue: c[r] = C[8g+r][n] inside each 16x16 subtile.
#pragma unroll
  for (int ni = 0; ni < 2; ++ni) {
    const int col = n0 + wn + 16 * ni + ln;
    const float bv = bias ? bias[col] : 0.f;
#pragma unroll
    for (int mi = 0; mi < 2; ++mi) {
#pragma unroll
      for (int r = 0; r < 8; ++r) {
        const size_t row = (size_t)(m0 + wm + 16 * mi + 8 * g + r);
        float v = acc[mi][ni][r] + bv;
        if (OM == 0) C[row * N + col] = v;
        if (OM == 1) Ch[row * N + col] = (_Float16)fmaxf(v, 0.f);
      }
    }
  }
}

// ---------------------------------------------------------------------------
// Attention: one block per (b,h); K/V head [50][32] staged in LDS, rows padded
// to 33 floats (gcd(33,64)=1 -> conflict-free). Wave w handles sq = w, w+8...
// Lane l owns feature dim d=l AND keys {l, l+32}. Scores: broadcast q one
// element at a time (32 unrolled __shfl), each lane accumulates its two keys'
// dots. Output written as f16 (only consumer is the Wo GEMM).
// ---------------------------------------------------------------------------
__global__ __launch_bounds__(256) void attention_k(
    const float* __restrict__ Qp, const float* __restrict__ Kp,
    const float* __restrict__ Vp, _Float16* __restrict__ Oh, int causal) {
  __shared__ float Ksh[S_ * 33];
  __shared__ float Vsh[S_ * 33];
  const int b = blockIdx.x >> 3;
  const int h = blockIdx.x & 7;
  const int tid = threadIdx.x;
  for (int idx = tid; idx < S_ * HD_; idx += 256) {
    int sk = idx >> 5, d = idx & 31;
    size_t off = ((size_t)sk * B_ + b) * D_ + h * HD_ + d;
    Ksh[sk * 33 + d] = Kp[off];
    Vsh[sk * 33 + d] = Vp[off];
  }
  __syncthreads();
  const int w = tid >> 5, l = tid & 31;
  const int l2 = (l + 32 < S_) ? (l + 32) : l;     // clamped second key row
  const float scale = 0.17677669529663687f;        // 1/sqrt(32)
  for (int sq = w; sq < S_; sq += 8) {
    size_t off = ((size_t)sq * B_ + b) * D_ + h * HD_ + l;
    float qv = Qp[off] * scale;                    // lane l holds q[d=l]
    const int kmax = causal ? (sq + 1) : S_;
    float a0 = 0.f, a1 = 0.f;                      // dots for keys l, l+32
#pragma unroll
    for (int d = 0; d < 32; ++d) {
      float qd = __shfl(qv, d, 32);
      a0 += qd * Ksh[l * 33 + d];
      a1 += qd * Ksh[l2 * 33 + d];
    }
    float sc0 = (l < kmax) ? a0 : -1e30f;
    float sc1 = (l + 32 < kmax) ? a1 : -1e30f;
    float m = fmaxf(sc0, sc1);
#pragma unroll
    for (int o = 16; o > 0; o >>= 1) m = fmaxf(m, __shfl_xor(m, o, 32));
    float e0 = __expf(sc0 - m), e1 = __expf(sc1 - m);
    float ssum = e0 + e1;
#pragma unroll
    for (int o = 16; o > 0; o >>= 1) ssum += __shfl_xor(ssum, o, 32);
    float od = 0.f;
    for (int sk = 0; sk < kmax; ++sk) {
      float ev = (sk < 32) ? __shfl(e0, sk, 32) : __shfl(e1, sk - 32, 32);
      od += ev * Vsh[sk * 33 + l];
    }
    Oh[off] = (_Float16)(od / ssum);
  }
}

// ---------------------------------------------------------------------------
// LayerNorm over D=256: one wave per token, 8 values per lane (b128 loads).
// MODE 0: O = LN(A + Rr)*g + beta, also writes f16 mirror O16.
// MODE 1: O  = 0.5*relu(LN(A)*g + beta)    (prefix stream 0, writes)
// MODE 2: O += 0.5*relu(LN(A)*g + beta)    (prefix stream 1, accumulates)
// MODE 3: O = LN(A)*g + beta               (final norm, f32 only)
// ---------------------------------------------------------------------------
__device__ inline void load8(const float* p, float* v) {
  float4 a = *(const float4*)p; float4 b = *(const float4*)(p + 4);
  v[0]=a.x; v[1]=a.y; v[2]=a.z; v[3]=a.w; v[4]=b.x; v[5]=b.y; v[6]=b.z; v[7]=b.w;
}
__device__ inline void store8(float* p, const float* v) {
  *(float4*)p       = make_float4(v[0], v[1], v[2], v[3]);
  *(float4*)(p + 4) = make_float4(v[4], v[5], v[6], v[7]);
}
__device__ inline void store8h(_Float16* p, const float* v) {
  h8v h;
#pragma unroll
  for (int i = 0; i < 8; ++i) h[i] = (_Float16)v[i];
  *(h8v*)p = h;
}

template <int MODE>
__global__ __launch_bounds__(256) void layernorm_k(
    const float* __restrict__ A, const float* __restrict__ Rr,
    const float* __restrict__ g, const float* __restrict__ beta,
    float* __restrict__ O, _Float16* __restrict__ O16) {
  const int w = threadIdx.x >> 5, l = threadIdx.x & 31;
  const size_t t = (size_t)blockIdx.x * 8 + w;
  float v[8];
  load8(A + t * D_ + l * 8, v);
  if (MODE == 0) {
    float r[8];
    load8(Rr + t * D_ + l * 8, r);
#pragma unroll
    for (int i = 0; i < 8; ++i) v[i] += r[i];
  }
  float s1 = 0.f, s2 = 0.f;
#pragma unroll
  for (int i = 0; i < 8; ++i) { s1 += v[i]; s2 += v[i] * v[i]; }
#pragma unroll
  for (int o = 16; o > 0; o >>= 1) {
    s1 += __shfl_xor(s1, o, 32);
    s2 += __shfl_xor(s2, o, 32);
  }
  const float mean = s1 * (1.f / D_);
  const float var  = s2 * (1.f / D_) - mean * mean;
  const float inv  = rsqrtf(var + 1e-5f);
  float gg[8], bb[8];
  load8(g + l * 8, gg);
  load8(beta + l * 8, bb);
  float y[8];
#pragma unroll
  for (int i = 0; i < 8; ++i) {
    y[i] = (v[i] - mean) * inv * gg[i] + bb[i];
    if (MODE == 1 || MODE == 2) y[i] = fmaxf(y[i], 0.f) * 0.5f;
  }
  float* o = O + t * D_ + l * 8;
  if (MODE == 2) {
    float prev[8];
    load8(o, prev);
#pragma unroll
    for (int i = 0; i < 8; ++i) y[i] += prev[i];
  }
  store8(o, y);
  if (MODE == 0) store8h(O16 + t * D_ + l * 8, y);
}

// Gather prefix embedding (f16): Eg[(s*B+b)*E + e] = emb[inp[b*S+s]*E + e]
__global__ __launch_bounds__(256) void gather_emb_k(
    const int* __restrict__ inp, const float* __restrict__ emb,
    _Float16* __restrict__ Eg) {
  const int s = blockIdx.y;
  const int j = blockIdx.x * 256 + threadIdx.x;   // j < B*E
  const int b = j >> 6, e = j & 63;
  const int tok = inp[b * S_ + s];
  Eg[((size_t)s * B_ + b) * E_ + e] = (_Float16)emb[(size_t)tok * E_ + e];
}

// X[s,b,:] = (s>=1 && s<tn[b] ? feat[s-1,b,:] : 0) + pos[s,:]; f32 + f16.
__global__ __launch_bounds__(256) void shift_pos_k(
    const float* __restrict__ feat, const float* __restrict__ pos,
    const int* __restrict__ tn, float* __restrict__ X,
    _Float16* __restrict__ X16) {
  const int s = blockIdx.y;
  const int j = blockIdx.x * 256 + threadIdx.x;   // j < B*D
  const int b = j >> 8, d = j & 255;
  float v = pos[(size_t)s * D_ + d];
  if (s >= 1 && s < tn[b]) v += feat[((size_t)(s - 1) * B_ + b) * D_ + d];
  const size_t o = ((size_t)s * B_ + b) * D_ + d;
  X[o] = v;
  X16[o] = (_Float16)v;
}

// fp32 -> f16 bulk convert (for z), 4 elements / thread.
__global__ __launch_bounds__(256) void cvt16_k(const float* __restrict__ in,
                                               _Float16* __restrict__ out) {
  const size_t j = ((size_t)blockIdx.x * 256 + threadIdx.x) * 4;
  float4 f = *(const float4*)(in + j);
  h4v h; h[0]=(_Float16)f.x; h[1]=(_Float16)f.y; h[2]=(_Float16)f.z; h[3]=(_Float16)f.w;
  *(h4v*)(out + j) = h;
}

// Weight transpose-convert: Wt[n*K + k] = (f16) W[k*N + n]  (read coalesced)
__global__ __launch_bounds__(256) void wtc_k(const float* __restrict__ W,
                                             _Float16* __restrict__ Wt,
                                             int K, int N) {
  const int idx = blockIdx.x * 256 + threadIdx.x;   // < K*N
  const int n = idx % N, k = idx / N;
  Wt[(size_t)n * K + k] = (_Float16)W[(size_t)k * N + n];
}

// ---------------------------------------------------------------------------
// Input leaf map. Top level in setup_inputs() insertion order:
//   0:src  1:z  2:text_num  3:font_inp  4:color_inp  5..:params leaves.
// params flattened JAX-pytree style (sorted dict keys at every level):
//   5:final_b 6:final_g
//   layers li (base 7+26*li): 0:W1 1:W2 2:b1 3:b2
//     4..11: cross {Wk,Wo,Wq,Wv,bk,bo,bq,bv}
//     12..17: n1b,n1g,n2b,n2g,n3b,n3g
//     18..25: self {Wk,Wo,Wq,Wv,bk,bo,bq,bv}
//   111: pos
//   prefix p (base 112+5*p): W,b,beta,emb,g
// ---------------------------------------------------------------------------
extern "C" void kernel_launch(void* const* d_in, const int* in_sizes, int n_in,
                              void* d_out, int out_size, void* d_ws,
                              size_t ws_size, hipStream_t stream) {
  (void)in_sizes; (void)n_in; (void)out_size; (void)ws_size;
  auto F = [&](int i) { return (const float*)d_in[i]; };
  const float* z        = F(1);
  const int*   text_num = (const int*)d_in[2];
  const int*   font     = (const int*)d_in[3];
  const int*   color    = (const int*)d_in[4];
  const float* final_b  = F(5);
  const float* final_g  = F(6);
  const float* pos      = F(111);

  // Workspace: fp32 X,R,Q,Kb,Vb,F1feat (6*[T,D]) then f16 region.
  float* ws = (float*)d_ws;
  const size_t TD = (size_t)T_ * D_;
  float* X   = ws;
  float* R   = ws + 1 * TD;
  float* Q   = ws + 2 * TD;
  float* Kb  = ws + 3 * TD;
  float* Vb  = ws + 4 * TD;
  float* F1  = ws + 5 * TD;                 // prefix feat [T,D] fp32
  _Float16* hb  = (_Float16*)(ws + 6 * TD);
  _Float16* X16 = hb;                       // [T,D]
  _Float16* Q16 = hb + TD;                  // [T,D] (also prefix Eg [T,E])
  _Float16* Z16 = hb + 2 * TD;              // [T,D]
  _Float16* Fh  = hb + 3 * TD;              // [T,FF] f16 FFN intermediate
  _Float16* wp  = hb + 3 * TD + (size_t)T_ * FF_;   // weight arena (~6.4MB)

  const dim3 blk(256);
  const dim3 gD(D_ / BN, T_ / BM);      // N=256 GEMM grid (4, 400)
  const dim3 gF(FF_ / BN, T_ / BM);     // N=512 GEMM grid (8, 400)
  const dim3 gLN(T_ / 8);

  // ---- one-time (per call) weight transpose-convert into f16 [N][K] ----
  auto conv = [&](const float* W, int K, int N) {
    _Float16* dst = wp;
    wtc_k<<<dim3((K * N) / 256), blk, 0, stream>>>(W, dst, K, N);
    wp += (size_t)K * N;
    return dst;
  };
  _Float16 *tsWq[L_], *tsWk[L_], *tsWv[L_], *tsWo[L_];
  _Float16 *tcWq[L_], *tcWk[L_], *tcWv[L_], *tcWo[L_];
  _Float16 *tW1[L_], *tW2[L_], *tPW[2];
  for (int li = 0; li < L_; ++li) {
    const int Lb = 7 + 26 * li;
    tsWq[li] = conv(F(Lb + 20), D_, D_);
    tsWk[li] = conv(F(Lb + 18), D_, D_);
    tsWv[li] = conv(F(Lb + 21), D_, D_);
    tsWo[li] = conv(F(Lb + 19), D_, D_);
    tcWq[li] = conv(F(Lb + 6),  D_, D_);
    tcWk[li] = conv(F(Lb + 4),  D_, D_);
    tcWv[li] = conv(F(Lb + 7),  D_, D_);
    tcWo[li] = conv(F(Lb + 5),  D_, D_);
    tW1[li]  = conv(F(Lb + 0),  D_, FF_);
    tW2[li]  = conv(F(Lb + 1),  FF_, D_);
  }
  tPW[0] = conv(F(112), E_, D_);
  tPW[1] = conv(F(117), E_, D_);
  cvt16_k<<<dim3(TD / 4 / 256), blk, 0, stream>>>(z, Z16);

  // ---- prefix: feat = 0.5*(relu(LN(emb0@W0+b0)) + relu(LN(emb1@W1+b1))) ----
  for (int p = 0; p < 2; ++p) {
    const int pb = 112 + 5 * p;
    const float* pbias = F(pb + 1), *pbeta = F(pb + 2), *pg = F(pb + 4);
    const int* inp = (p == 0) ? font : color;
    gather_emb_k<<<dim3((B_ * E_) / 256, S_), blk, 0, stream>>>(inp, F(pb + 3), Q16);
    gemm_wmma_k<0><<<gD, blk, 0, stream>>>(Q16, tPW[p], pbias, R, nullptr, T_, D_, E_);
    if (p == 0)
      layernorm_k<1><<<gLN, blk, 0, stream>>>(R, nullptr, pg, pbeta, F1, nullptr);
    else
      layernorm_k<2><<<gLN, blk, 0, stream>>>(R, nullptr, pg, pbeta, F1, nullptr);
  }
  shift_pos_k<<<dim3((B_ * D_) / 256, S_), blk, 0, stream>>>(F1, pos, text_num, X, X16);

  // ---- decoder layers (post-norm) ----
  for (int li = 0; li < L_; ++li) {
    const int Lb = 7 + 26 * li;
    const float *b1 = F(Lb + 2), *b2 = F(Lb + 3);
    const float *cbk = F(Lb + 8), *cbo = F(Lb + 9), *cbq = F(Lb + 10), *cbv = F(Lb + 11);
    const float *n1b = F(Lb + 12), *n1g = F(Lb + 13);
    const float *n2b = F(Lb + 14), *n2g = F(Lb + 15);
    const float *n3b = F(Lb + 16), *n3g = F(Lb + 17);
    const float *sbk = F(Lb + 22), *sbo = F(Lb + 23), *sbq = F(Lb + 24), *sbv = F(Lb + 25);

    // self-attention (causal)
    gemm_wmma_k<0><<<gD, blk, 0, stream>>>(X16, tsWq[li], sbq, Q,  nullptr, T_, D_, D_);
    gemm_wmma_k<0><<<gD, blk, 0, stream>>>(X16, tsWk[li], sbk, Kb, nullptr, T_, D_, D_);
    gemm_wmma_k<0><<<gD, blk, 0, stream>>>(X16, tsWv[li], sbv, Vb, nullptr, T_, D_, D_);
    attention_k<<<B_ * H_, blk, 0, stream>>>(Q, Kb, Vb, Q16, 1);
    gemm_wmma_k<0><<<gD, blk, 0, stream>>>(Q16, tsWo[li], sbo, R, nullptr, T_, D_, D_);
    layernorm_k<0><<<gLN, blk, 0, stream>>>(X, R, n1g, n1b, X, X16);
    // cross-attention over z
    gemm_wmma_k<0><<<gD, blk, 0, stream>>>(X16, tcWq[li], cbq, Q,  nullptr, T_, D_, D_);
    gemm_wmma_k<0><<<gD, blk, 0, stream>>>(Z16, tcWk[li], cbk, Kb, nullptr, T_, D_, D_);
    gemm_wmma_k<0><<<gD, blk, 0, stream>>>(Z16, tcWv[li], cbv, Vb, nullptr, T_, D_, D_);
    attention_k<<<B_ * H_, blk, 0, stream>>>(Q, Kb, Vb, Q16, 0);
    gemm_wmma_k<0><<<gD, blk, 0, stream>>>(Q16, tcWo[li], cbo, R, nullptr, T_, D_, D_);
    layernorm_k<0><<<gLN, blk, 0, stream>>>(X, R, n2g, n2b, X, X16);
    // FFN (hidden kept f16-only: sole consumer is FFN2 GEMM)
    gemm_wmma_k<1><<<gF, blk, 0, stream>>>(X16, tW1[li], b1, nullptr, Fh, T_, FF_, D_);
    gemm_wmma_k<0><<<gD, blk, 0, stream>>>(Fh, tW2[li], b2, R, nullptr, T_, D_, FF_);
    layernorm_k<0><<<gLN, blk, 0, stream>>>(X, R, n3g, n3b, X, X16);
  }

  // final LN -> d_out ([S,B,D] fp32)
  layernorm_k<3><<<gLN, blk, 0, stream>>>(X, nullptr, final_g, final_b,
                                          (float*)d_out, nullptr);
}
